// SparseCADGCN_26414048871029
// MI455X (gfx1250) — compile-verified, matrix operand order checked by reading, not run on hardware
//
#include <hip/hip_runtime.h>
#include <hip/hip_bf16.h>

typedef __attribute__((ext_vector_type(16))) __bf16 v16bf;
typedef __attribute__((ext_vector_type(8)))  float  v8f;

#define CHN 128

static __device__ __forceinline__ unsigned short f2bf(float f) {
  union { float f; unsigned u; } v; v.f = f;
  return (unsigned short)((v.u + 0x7FFFu + ((v.u >> 16) & 1u)) >> 16);
}

// ---------------- weight prep: Wt[n*Kpad+k] = bf16(W[k*N+n]) ----------------
__global__ void k_prep_weight(const float* __restrict__ W, unsigned short* __restrict__ Wt,
                              int K, int N, int Kpad) {
  int i = blockIdx.x * blockDim.x + threadIdx.x;
  if (i >= N * Kpad) return;
  int n = i / Kpad, k = i % Kpad;
  float v = (k < K) ? W[k * N + n] : 0.0f;
  Wt[i] = f2bf(v);
}

__global__ void k_cast_bf16(const float* __restrict__ in, unsigned short* __restrict__ out, int n) {
  int i = blockIdx.x * blockDim.x + threadIdx.x;
  if (i < n) out[i] = f2bf(in[i]);
}

__global__ void k_count(const int* __restrict__ idx, int* __restrict__ cnt, int n) {
  int i = blockIdx.x * blockDim.x + threadIdx.x;
  if (i < n) atomicAdd(&cnt[idx[i]], 1);
}

// ---------------- fused edge-message MLP + scatter (max / sum) ----------------
// 16 edges per block. 512 threads = 16 waves: waves 0-7 -> node stream (W, atomic max),
// waves 8-15 -> super stream (Ws, atomic add). Wcat: [256][K_PAD] bf16 (cols 0-127 = W^T,
// 128-255 = Ws^T).
template<int CH_IN, int K_PAD>
__global__ void __launch_bounds__(512)
k_edge_conv(const float* __restrict__ xN, int ldN,
            const float* __restrict__ xS, int ldS,
            const float* __restrict__ eattr,
            const int* __restrict__ src, const int* __restrict__ dst,
            const unsigned short* __restrict__ Wcat,
            const float* __restrict__ bN, const float* __restrict__ bS,
            int* __restrict__ fmax, float* __restrict__ ssum, int n_edges) {
  __shared__ __align__(64) unsigned short A[2][16 * K_PAD];
  __shared__ int sE[16], dE[16];
  const int tid = threadIdx.x;
  const int e0  = blockIdx.x * 16;
  if (tid < 16) {
    int e = e0 + tid;
    sE[tid] = (e < n_edges) ? src[e] : 0;
    dE[tid] = (e < n_edges) ? dst[e] : -1;
  }
  __syncthreads();
  for (int i = tid; i < 16 * K_PAD; i += 512) {
    int el = i / K_PAD, k = i % K_PAD;
    float vn = 0.f, vs = 0.f;
    if (dE[el] >= 0) {
      int s = sE[el], d = dE[el];
      if (k < CH_IN)            { vn = xN[s*ldN + k];                    vs = xS[s*ldS + k]; }
      else if (k < 2*CH_IN)     { int kk = k - CH_IN;
                                  vn = xN[d*ldN + kk] - xN[s*ldN + kk];
                                  vs = xS[d*ldS + kk] - xS[s*ldS + kk]; }
      else if (k < 2*CH_IN + 6) { float ea = eattr[(e0+el)*6 + (k - 2*CH_IN)]; vn = ea; vs = ea; }
    }
    A[0][i] = f2bf(vn);
    A[1][i] = f2bf(vs);
  }
  __syncthreads();
  const int lane   = tid & 31;
  const int wave   = tid >> 5;        // 0..15
  const int stream = wave >> 3;       // 0 = max stream, 1 = mean stream
  const int colg   = wave * 16 + (lane & 15);   // 0..255 into Wcat
  const int khalf  = (lane >> 4) << 4;
  const unsigned short* Arow = &A[stream][(lane & 15) * K_PAD + khalf];
  const unsigned short* Brow = &Wcat[colg * K_PAD + khalf];
  v8f acc = {};
#pragma unroll
  for (int kb = 0; kb < K_PAD; kb += 32) {
    v16bf a = *(const v16bf*)(Arow + kb);
    v16bf b = *(const v16bf*)(Brow + kb);
    acc = __builtin_amdgcn_wmma_f32_16x16x32_bf16(false, a, false, b, (short)0, acc, false, false);
  }
  const int   outc = colg & (CHN - 1);
  const float bias = stream ? bS[outc] : bN[outc];
#pragma unroll
  for (int j = 0; j < 8; ++j) {
    int m = j + ((lane >> 4) << 3);   // edge within tile
    int d = dE[m];
    if (d < 0) continue;
    float v = fmaxf(acc[j] + bias, 0.f);      // ReLU -> value >= 0
    if (stream) atomicAdd(&ssum[d * CHN + outc], v);
    else        atomicMax(&fmax[d * CHN + outc], __float_as_int(v));
  }
}

// ---------------- generic WMMA GEMM: C = act(A[M,K] @ Bt[N,K]^T + bias) ----------------
// optional scatter-max epilogue (bbox pooling). 256 threads = 8 waves, each a 16-col tile.
__global__ void __launch_bounds__(256)
k_gemm(const unsigned short* __restrict__ A, const unsigned short* __restrict__ Bt,
       const float* __restrict__ bias,
       float* __restrict__ Cf, unsigned short* __restrict__ Cb, int ldc, int coff,
       int M, int N, int K, int relu,
       const int* __restrict__ scat_idx, int* __restrict__ scat_bits, int scat_ld, int scat_off) {
  const int lane = threadIdx.x & 31;
  const int wave = threadIdx.x >> 5;
  const int nt   = blockIdx.y * 8 + wave;
  if (nt * 16 >= N) return;                    // wave-uniform exit
  const int col   = nt * 16 + (lane & 15);
  const int khalf = (lane >> 4) << 4;
  int rowA = blockIdx.x * 16 + (lane & 15);
  if (rowA >= M) rowA = M - 1;                 // clamp, mask on store
  const unsigned short* Ap = A + (size_t)rowA * K + khalf;
  const unsigned short* Bp = Bt + (size_t)col  * K + khalf;
  v8f acc = {};
  for (int kb = 0; kb < K; kb += 32) {
    v16bf a = *(const v16bf*)(Ap + kb);
    v16bf b = *(const v16bf*)(Bp + kb);
    acc = __builtin_amdgcn_wmma_f32_16x16x32_bf16(false, a, false, b, (short)0, acc, false, false);
  }
  const float bv = bias ? bias[col] : 0.f;
#pragma unroll
  for (int j = 0; j < 8; ++j) {
    int r = blockIdx.x * 16 + j + ((lane >> 4) << 3);
    if (r >= M) continue;
    float v = acc[j] + bv;
    if (relu) v = fmaxf(v, 0.f);
    if (scat_idx) {
      int bb = scat_idx[r];
      atomicMax(&scat_bits[bb * scat_ld + scat_off + col], __float_as_int(v));
    } else {
      if (Cf) Cf[(size_t)r * ldc + coff + col] = v;
      if (Cb) Cb[(size_t)r * ldc + coff + col] = f2bf(v);
    }
  }
}

// ---------------- per-node conv finish: residual add + mean division ----------------
__global__ void k_finish_conv(const int* __restrict__ fmax, const float* __restrict__ ssum,
                              const int* __restrict__ deg,
                              const float* __restrict__ prevN, const float* __restrict__ prevS,
                              float* __restrict__ outN, float* __restrict__ outS,
                              int ld, int nn) {
  int i = blockIdx.x * blockDim.x + threadIdx.x;
  if (i >= nn * CHN) return;
  int n = i >> 7, c = i & (CHN - 1);
  float g = __int_as_float(fmax[i]);              // bits of non-negative float
  int   d = deg[n];
  float s = ssum[i] / (float)(d > 1 ? d : 1);
  if (prevN) g += prevN[n * ld + c];
  if (prevS) s += prevS[n * ld + c];
  outN[n * ld + c] = g;
  outS[n * ld + c] = s;
}

// scatter feats into pooled[:,1024:1408] (max) and feats_super into bbox sums
__global__ void k_scatter_node(const float* __restrict__ feats, const float* __restrict__ featsS,
                               const int* __restrict__ bidx,
                               int* __restrict__ pooled, float* __restrict__ ssuper, int nn) {
  int i = blockIdx.x * blockDim.x + threadIdx.x;
  if (i >= nn * 384) return;
  int n = i / 384, c = i % 384;
  int bb = bidx[n];
  atomicMax(&pooled[bb * 1408 + 1024 + c], __float_as_int(feats[i]));
  atomicAdd(&ssuper[bb * 384 + c], featsS[i]);
}

__global__ void k_finish_bbox(const float* __restrict__ ssuper, const int* __restrict__ bcnt,
                              unsigned short* __restrict__ msbf, float* __restrict__ h0, int B) {
  int i = blockIdx.x * blockDim.x + threadIdx.x;
  if (i >= B * 384) return;
  int b = i / 384, c = i % 384;
  int cnt = bcnt[b];
  float m = ssuper[i] / (float)(cnt > 1 ? cnt : 1);
  msbf[i] = f2bf(m);
  h0[b * 2816 + 2432 + c] = m;
}

__global__ void k_pooled_copy(const int* __restrict__ pooled, float* __restrict__ h0, int B) {
  int i = blockIdx.x * blockDim.x + threadIdx.x;
  if (i >= B * 1408) return;
  int b = i / 1408, c = i % 1408;
  h0[b * 2816 + c] = __int_as_float(pooled[i]);
}

// =====================================================================================
extern "C" void kernel_launch(void* const* d_in, const int* in_sizes, int n_in,
                              void* d_out, int out_size, void* d_ws, size_t ws_size,
                              hipStream_t stream) {
  const float* x     = (const float*)d_in[0];
  const float* eattr = (const float*)d_in[1];
  const float* W_h   = (const float*)d_in[2];
  const float* b_h   = (const float*)d_in[3];
  const float* Ws_h  = (const float*)d_in[4];
  const float* bs_h  = (const float*)d_in[5];
  const float* Wb    = (const float*)d_in[6];
  const float* bb    = (const float*)d_in[7];
  const float* Wbs   = (const float*)d_in[8];
  const float* bbs   = (const float*)d_in[9];
  const float* W_f   = (const float*)d_in[10];
  const float* b_f   = (const float*)d_in[11];
  const float* W_fs  = (const float*)d_in[12];
  const float* b_fs  = (const float*)d_in[13];
  const float* W1    = (const float*)d_in[14];
  const float* b1    = (const float*)d_in[15];
  const float* W2    = (const float*)d_in[16];
  const float* b2    = (const float*)d_in[17];
  const float* W3    = (const float*)d_in[18];
  const float* b3    = (const float*)d_in[19];
  const int*   edge  = (const int*)d_in[20];
  const int*   bidx  = (const int*)d_in[21];

  const int NN = in_sizes[0] / 8;       // 50000 nodes
  const int NE = in_sizes[1] / 6;       // 800000 edges
  const int NB = 2000;                  // bboxes
  const int*   src = edge;
  const int*   dst = edge + NE;

  // ---- carve workspace ----
  char* p = (char*)d_ws;
  auto alloc = [&](size_t bytes) -> void* {
    void* r = (void*)p; p += (bytes + 255) & ~(size_t)255; return r;
  };
  unsigned short* wcat_h  = (unsigned short*)alloc((size_t)256 * 32  * 2);
  unsigned short* wcat_b0 = (unsigned short*)alloc((size_t)256 * 288 * 2);
  unsigned short* wcat_b1 = (unsigned short*)alloc((size_t)256 * 288 * 2);
  unsigned short* wft     = (unsigned short*)alloc((size_t)1024 * 384 * 2);
  unsigned short* wfst    = (unsigned short*)alloc((size_t)1024 * 384 * 2);
  unsigned short* w1t     = (unsigned short*)alloc((size_t)512 * 2816 * 2);
  unsigned short* w2t     = (unsigned short*)alloc((size_t)256 * 512 * 2);
  unsigned short* w3t     = (unsigned short*)alloc((size_t)32  * 256 * 2);
  int*            fmax    = (int*)  alloc((size_t)NN * CHN * 4);
  float*          ssum    = (float*)alloc((size_t)NN * CHN * 4);
  float*          feats   = (float*)alloc((size_t)NN * 384 * 4);
  float*          featsS  = (float*)alloc((size_t)NN * 384 * 4);
  unsigned short* featsbf = (unsigned short*)alloc((size_t)NN * 384 * 2);
  int*            deg     = (int*)  alloc((size_t)NN * 4);
  int*            bcnt    = (int*)  alloc((size_t)NB * 4);
  float*          ssuper  = (float*)alloc((size_t)NB * 384 * 4);
  int*            pooled  = (int*)  alloc((size_t)NB * 1408 * 4);
  unsigned short* msbf    = (unsigned short*)alloc((size_t)NB * 384 * 2);
  float*          h0      = (float*)alloc((size_t)NB * 2816 * 4);
  unsigned short* h0bf    = (unsigned short*)alloc((size_t)NB * 2816 * 2);
  unsigned short* h1bf    = (unsigned short*)alloc((size_t)NB * 512 * 2);
  unsigned short* h2bf    = (unsigned short*)alloc((size_t)NB * 256 * 2);

  auto prep = [&](const float* W, unsigned short* Wt, int K, int N, int Kpad) {
    int tot = N * Kpad;
    k_prep_weight<<<(tot + 255) / 256, 256, 0, stream>>>(W, Wt, K, N, Kpad);
  };
  // ---- weight transpose/cast ----
  prep(W_h,  wcat_h,            22, 128, 32);
  prep(Ws_h, wcat_h + 128 * 32, 22, 128, 32);
  prep(Wb,                wcat_b0,             262, 128, 288);
  prep(Wbs,               wcat_b0 + 128 * 288, 262, 128, 288);
  prep(Wb + 262 * 128,    wcat_b1,             262, 128, 288);
  prep(Wbs + 262 * 128,   wcat_b1 + 128 * 288, 262, 128, 288);
  prep(W_f,  wft,  384, 1024, 384);
  prep(W_fs, wfst, 384, 1024, 384);
  prep(W1, w1t, 2816, 512, 2816);
  prep(W2, w2t, 512, 256, 512);
  prep(W3, w3t, 256, 32, 256);

  // ---- counts + zero init ----
  hipMemsetAsync(deg,    0, (size_t)NN * 4, stream);
  hipMemsetAsync(bcnt,   0, (size_t)NB * 4, stream);
  hipMemsetAsync(ssuper, 0, (size_t)NB * 384 * 4, stream);
  hipMemsetAsync(pooled, 0, (size_t)NB * 1408 * 4, stream);
  k_count<<<(NE + 255) / 256, 256, 0, stream>>>(dst,  deg,  NE);
  k_count<<<(NN + 255) / 256, 256, 0, stream>>>(bidx, bcnt, NN);

  const int etiles = (NE + 15) / 16;
  // ---- head conv ----
  hipMemsetAsync(fmax, 0, (size_t)NN * CHN * 4, stream);
  hipMemsetAsync(ssum, 0, (size_t)NN * CHN * 4, stream);
  k_edge_conv<8, 32><<<etiles, 512, 0, stream>>>(x, 8, x, 8, eattr, src, dst,
      wcat_h, b_h, bs_h, fmax, ssum, NE);
  k_finish_conv<<<(NN * CHN + 255) / 256, 256, 0, stream>>>(fmax, ssum, deg,
      nullptr, nullptr, feats, featsS, 384, NN);
  // ---- residual blocks ----
  for (int blk = 0; blk < 2; ++blk) {
    hipMemsetAsync(fmax, 0, (size_t)NN * CHN * 4, stream);
    hipMemsetAsync(ssum, 0, (size_t)NN * CHN * 4, stream);
    const unsigned short* wc = blk ? wcat_b1 : wcat_b0;
    k_edge_conv<128, 288><<<etiles, 512, 0, stream>>>(
        feats + blk * CHN, 384, featsS + blk * CHN, 384, eattr, src, dst,
        wc, bb + blk * CHN, bbs + blk * CHN, fmax, ssum, NE);
    k_finish_conv<<<(NN * CHN + 255) / 256, 256, 0, stream>>>(fmax, ssum, deg,
        feats + blk * CHN, featsS + blk * CHN,
        feats + (blk + 1) * CHN, featsS + (blk + 1) * CHN, 384, NN);
  }
  // ---- pooling scatters + node fusion (fused atomic-max pooling epilogue) ----
  k_cast_bf16<<<(NN * 384 + 255) / 256, 256, 0, stream>>>(feats, featsbf, NN * 384);
  k_scatter_node<<<(NN * 384 + 255) / 256, 256, 0, stream>>>(feats, featsS, bidx,
      pooled, ssuper, NN);
  {
    dim3 g((NN + 15) / 16, 1024 / 128);
    k_gemm<<<g, 256, 0, stream>>>(featsbf, wft, b_f, nullptr, nullptr, 0, 0,
        NN, 1024, 384, 1, bidx, pooled, 1408, 0);
  }
  // ---- bbox mean + super fusion ----
  k_finish_bbox<<<(NB * 384 + 255) / 256, 256, 0, stream>>>(ssuper, bcnt, msbf, h0, NB);
  {
    dim3 g((NB + 15) / 16, 1024 / 128);
    k_gemm<<<g, 256, 0, stream>>>(msbf, wfst, b_fs, h0, nullptr, 2816, 1408,
        NB, 1024, 384, 1, nullptr, nullptr, 0, 0);
  }
  k_pooled_copy<<<(NB * 1408 + 255) / 256, 256, 0, stream>>>(pooled, h0, NB);
  k_cast_bf16<<<(NB * 2816 + 255) / 256, 256, 0, stream>>>(h0, h0bf, NB * 2816);
  // ---- classifier MLP ----
  {
    dim3 g((NB + 15) / 16, 4);   // 512 cols
    k_gemm<<<g, 256, 0, stream>>>(h0bf, w1t, b1, nullptr, h1bf, 512, 0,
        NB, 512, 2816, 1, nullptr, nullptr, 0, 0);
  }
  {
    dim3 g((NB + 15) / 16, 2);   // 256 cols
    k_gemm<<<g, 256, 0, stream>>>(h1bf, w2t, b2, nullptr, h2bf, 256, 0,
        NB, 256, 512, 1, nullptr, nullptr, 0, 0);
  }
  {
    dim3 g((NB + 15) / 16, 1);   // 32 cols
    k_gemm<<<g, 256, 0, stream>>>(h2bf, w3t, b3, (float*)d_out, nullptr, 32, 0,
        NB, 32, 256, 0, nullptr, nullptr, 0, 0);
  }
}
